// VariateEncoder_60181081752051
// MI455X (gfx1250) — matrix operand order repaired and checked by reading, not compiled, not run on hardware
//
#include <hip/hip_runtime.h>

// ---------------------------------------------------------------------------
// Mamba-style SSM block with L=1 (x reshaped to (B*V, 1, D_MODEL)):
//   * conv reduces to the last tap:  xc = silu(xs*conv_w[3] + conv_b)
//   * the scan is one step from h0=0:  h = dBx  (dA / A_log are irrelevant)
//   * y[d] = xc[d] * (dt[d]*dot(B,C) + D[d]), then y *= silu(gate)
// Each layer = LN + 4 GEMMs + elementwise, fused into one kernel.
// GEMMs: v_wmma_f32_16x16x32_f16 (wave32, 16 tokens/wave). GEMM1's B-panels
// are double-buffered in LDS via global_load_async_to_lds_b128 + ASYNCcnt.
// ---------------------------------------------------------------------------

typedef __attribute__((ext_vector_type(16))) _Float16 v16h;
typedef __attribute__((ext_vector_type(8)))  float    v8f;

#define D_MODEL  256
#define D_INNER  512
#define D_STATE  64
#define DT_RANK  16
#define N_TOK    3200      // B*V = 8*400
#define N_LAYERS 2

// Packed f16 weight geometry (units: halves). One B-tile = 32 lanes x 16
// halves = 512 halves = 1024 bytes, laid out exactly as the WMMA B operand.
#define PW_INW_OFF    0                         // in_w:   8 kt x 64 nt tiles
#define PW_XPROJ_OFF  (PW_INW_OFF   + 8*64*512) // xproj: 16 kt x  9 nt tiles
#define PW_DTPROJ_OFF (PW_XPROJ_OFF + 16*9*512) // dtproj: 1 kt x 32 nt tiles
#define PW_OUTW_OFF   (PW_DTPROJ_OFF + 1*32*512)// out_w: 16 kt x 16 nt tiles
#define PW_LAYER_HALVES (PW_OUTW_OFF + 16*16*512)

#define TILES_INW   (8*64)
#define TILES_XPROJ (16*9)
#define TILES_DT    (1*32)
#define TILES_OUTW  (16*16)
#define TILES_PER_LAYER (TILES_INW + TILES_XPROJ + TILES_DT + TILES_OUTW) // 944

// ---------------------------------------------------------------------------
// CDNA5 async global->LDS copy (16B per lane) + ASYNCcnt gates (inline asm;
// the clang builtin arity for this family is toolchain-dependent).
// LDS byte offset = low 32 bits of the generic pointer (ISA 10.2: flat LDS
// address is {SHARED_BASE, offset[31:0]}).
// ---------------------------------------------------------------------------
__device__ __forceinline__ unsigned lds_off32(const void* p) {
  return (unsigned)(unsigned long long)p;
}
__device__ __forceinline__ void async_ld16(unsigned ldsoff, const void* g) {
  asm volatile("global_load_async_to_lds_b128 %0, %1, off"
               :: "v"(ldsoff), "v"(g) : "memory");
}
__device__ __forceinline__ void wait_async0()  {
  asm volatile("s_wait_asynccnt 0" ::: "memory");
}
__device__ __forceinline__ void wait_async16() {
  asm volatile("s_wait_asynccnt 16" ::: "memory");
}

// ---------------------------------------------------------------------------
// Pack all weight matrices (both layers) f32 -> f16 in WMMA-B tile layout:
// tile (kt,nt), lane l (n=l&15, g=l>>4), half i:  K = kt*32 + (i<8 ? i : i+8) + 8*g,
// N = nt*16 + n.  Out-of-range K/N zero-filled (pads DT_RANK 16->32 and
// xproj's N=144 -> 9 tiles).
// ---------------------------------------------------------------------------
__global__ void pack_weights(const float* __restrict__ in_w,
                             const float* __restrict__ xproj_w,
                             const float* __restrict__ dtproj_w,
                             const float* __restrict__ out_w,
                             _Float16*    __restrict__ pw) {
  int unit = blockIdx.x * blockDim.x + threadIdx.x;   // (layer, tile, lane)
  int total = N_LAYERS * TILES_PER_LAYER * 32;
  if (unit >= total) return;

  int lane  = unit & 31;
  int t     = unit >> 5;
  int layer = t / TILES_PER_LAYER;
  int rt    = t % TILES_PER_LAYER;

  const float* src; int K, N, ntiles; unsigned dstBase; int tile;
  if (rt < TILES_INW) {
    tile = rt;
    src = in_w + (size_t)layer * D_MODEL * (2*D_INNER);
    K = D_MODEL; N = 2*D_INNER; ntiles = 64; dstBase = PW_INW_OFF;
  } else if (rt < TILES_INW + TILES_XPROJ) {
    tile = rt - TILES_INW;
    src = xproj_w + (size_t)layer * D_INNER * (DT_RANK + 2*D_STATE);
    K = D_INNER; N = DT_RANK + 2*D_STATE; ntiles = 9; dstBase = PW_XPROJ_OFF;
  } else if (rt < TILES_INW + TILES_XPROJ + TILES_DT) {
    tile = rt - (TILES_INW + TILES_XPROJ);
    src = dtproj_w + (size_t)layer * DT_RANK * D_INNER;
    K = DT_RANK; N = D_INNER; ntiles = 32; dstBase = PW_DTPROJ_OFF;
  } else {
    tile = rt - (TILES_INW + TILES_XPROJ + TILES_DT);
    src = out_w + (size_t)layer * D_INNER * D_MODEL;
    K = D_INNER; N = D_MODEL; ntiles = 16; dstBase = PW_OUTW_OFF;
  }

  int kt = tile / ntiles, nt = tile % ntiles;
  int n = lane & 15, g = lane >> 4;
  _Float16* dst = pw + (size_t)layer * PW_LAYER_HALVES + dstBase
                     + ((unsigned)tile * 32u + (unsigned)lane) * 16u;
#pragma unroll
  for (int i = 0; i < 16; i++) {
    int Krow = kt*32 + (i < 8 ? i : i + 8) + 8*g;
    int Ncol = nt*16 + n;
    float v = (Krow < K && Ncol < N) ? src[(size_t)Krow * N + Ncol] : 0.f;
    dst[i] = (_Float16)v;
  }
}

__device__ __forceinline__ float silu_f(float v) {
  return v * (1.f / (1.f + __expf(-v)));
}

// ---------------------------------------------------------------------------
// Fused SSM layer. 1 wave (32 threads) per 16-token tile; grid = 200 blocks.
// Weight pointers are pre-offset to this layer. A_log unused (L=1).
// ---------------------------------------------------------------------------
__global__ void __launch_bounds__(32)
ssm_layer(const float*    __restrict__ xin,    // [N_TOK, 256]
          float*          __restrict__ xout,   // [N_TOK, 256]
          const _Float16* __restrict__ pw,     // packed weights, this layer
          const float* __restrict__ ln_w, const float* __restrict__ ln_b,
          const float* __restrict__ in_b,      // [1024]
          const float* __restrict__ conv_w,    // [4,512]   (only row 3 used)
          const float* __restrict__ conv_b,    // [512]
          const float* __restrict__ dt_b,      // [512]
          const float* __restrict__ dpar,      // [512]
          const float* __restrict__ out_b)     // [256]
{
  __shared__ __align__(32) _Float16 s_xc[16 * D_INNER];  // silu(conv(xs))
  __shared__ __align__(32) _Float16 s_gy[16 * D_INNER];  // gate, later y
  __shared__ __align__(32) _Float16 s_wpanel[2 * 8 * 512]; // GEMM1 B panels
  __shared__ __align__(32) _Float16 s_dtr[16 * 16];       // dt_raw tile
  __shared__ float s_mu[16], s_rs[16];

  const int lane = threadIdx.x;             // 0..31
  const int tok0 = blockIdx.x * 16;
  const int m16  = lane & 15;               // N-column / M-row index
  const int g    = lane >> 4;

  // ---- LayerNorm statistics: 2 lanes per token, 128 elems each ------------
  {
    int t = lane >> 1, hh = lane & 1;
    const float* row = xin + (size_t)(tok0 + t) * D_MODEL + hh * 128;
    float s = 0.f, s2 = 0.f;
    for (int j = 0; j < 128; j++) { float v = row[j]; s += v; s2 += v * v; }
    s  += __shfl_xor(s, 1);
    s2 += __shfl_xor(s2, 1);
    float mu  = s * (1.f / 256.f);
    float var = s2 * (1.f / 256.f) - mu * mu;
    if (hh == 0) { s_mu[t] = mu; s_rs[t] = rsqrtf(var + 1e-5f); }
  }
  __syncthreads();

  // ---- Normalized A-tiles for GEMM1 (ISA 16-bit A 16x32 layout) -----------
  // lane l (m=l&15, g=l>>4), half i: K = (i<8 ? i : i+8) + 8*g
  v16h a1[8];
  {
    float mu = s_mu[m16], rs = s_rs[m16];
    const float* row = xin + (size_t)(tok0 + m16) * D_MODEL;
#pragma unroll
    for (int kt = 0; kt < 8; kt++)
#pragma unroll
      for (int i = 0; i < 16; i++) {
        int K = kt*32 + (i < 8 ? i : i + 8) + 8*g;
        a1[kt][i] = (_Float16)((row[K] - mu) * rs * ln_w[K] + ln_b[K]);
      }
  }

  // ---- GEMM1: [16x256] @ [256x1024] + in_b; conv tap + silu fused ---------
  // B panels (8 kt-tiles = 8KB) double-buffered in LDS via async copies:
  // prefetch panel nt+1 while computing panel nt; s_wait_asynccnt 16 leaves
  // exactly the next panel (16 ops) in flight => current panel complete.
  const _Float16* pw_in = pw + PW_INW_OFF;
  const unsigned wp_base = lds_off32(s_wpanel) + (unsigned)lane * 32u;

  auto issue_panel = [&](int nt, int slot) {
#pragma unroll
    for (int kt = 0; kt < 8; kt++) {
      const _Float16* src = pw_in + ((kt*64 + nt)*32 + lane) * 16;
      unsigned dst = wp_base + (unsigned)slot * 8192u + (unsigned)kt * 1024u;
      async_ld16(dst,      src);
      async_ld16(dst + 16, src + 8);
    }
  };

  issue_panel(0, 0);
  for (int nt = 0; nt < 64; nt++) {
    int slot = nt & 1;
    if (nt < 63) { issue_panel(nt + 1, slot ^ 1); wait_async16(); }
    else         { wait_async0(); }

    float bias = in_b[nt*16 + m16];
    v8f acc; for (int r = 0; r < 8; r++) acc[r] = bias;
    const _Float16* bp = s_wpanel + slot * (8*512) + lane * 16;
#pragma unroll
    for (int kt = 0; kt < 8; kt++) {
      v16h b = *(const v16h*)(bp + kt * 512);
      acc = __builtin_amdgcn_wmma_f32_16x16x32_f16(false, a1[kt], false, b,
                                                   (short)0, acc, false, false);
    }
    if (nt < 32) {                       // xs half -> xc = silu(xs*cw3 + cb)
      int d = nt*16 + m16;
      float cw = conv_w[3*D_INNER + d], cb = conv_b[d];
      for (int r = 0; r < 8; r++)
        s_xc[(r + 8*g) * D_INNER + d] = (_Float16)silu_f(acc[r]*cw + cb);
    } else {                             // gate half (raw; silu applied later)
      int d = (nt - 32)*16 + m16;
      for (int r = 0; r < 8; r++)
        s_gy[(r + 8*g) * D_INNER + d] = (_Float16)acc[r];
    }
  }
  __syncthreads();

  // A-tile loader from row-major [16 x 512] f16 LDS: half-pairs (i,i+1) are
  // consecutive K, so each tile is 8 aligned dword loads.
  auto loadA = [&](const _Float16* sm, int kt) -> v16h {
    union { v16h h; unsigned u[8]; } A;
    const unsigned* base = (const unsigned*)(sm + m16 * D_INNER + kt * 32);
#pragma unroll
    for (int p = 0; p < 8; p++) {
      int dw = (p < 4 ? p : 4 + p) + 4*g;   // {0..3,8..11} + 4g
      A.u[p] = base[dw];
    }
    return A.h;
  };

  // ---- GEMM2: xc[16x512] @ xproj[512x144] -> dt_raw | B | C ---------------
  const _Float16* pw_xp = pw + PW_XPROJ_OFF;
  v8f bm[4];
  v8f pbc; for (int r = 0; r < 8; r++) pbc[r] = 0.f;
  for (int nt = 0; nt < 9; nt++) {
    v8f acc; for (int r = 0; r < 8; r++) acc[r] = 0.f;
    for (int kt = 0; kt < 16; kt++) {
      v16h a = loadA(s_xc, kt);
      v16h b = *(const v16h*)(pw_xp + ((kt*9 + nt)*32 + lane) * 16);
      acc = __builtin_amdgcn_wmma_f32_16x16x32_f16(false, a, false, b,
                                                   (short)0, acc, false, false);
    }
    if (nt == 0) {                       // dt_raw (16x16)
      for (int r = 0; r < 8; r++)
        s_dtr[(r + 8*g) * 16 + m16] = (_Float16)acc[r];
    } else if (nt <= 4) {                // B tiles (s = (nt-1)*16 + n)
      bm[nt - 1] = acc;
    } else {                             // C tiles: accumulate B.C elementwise
      for (int r = 0; r < 8; r++) pbc[r] += bm[nt - 5][r] * acc[r];
    }
  }
  // bc[m] = dot(B,C): reduce over the 16 lanes of each half-wave
  for (int r = 0; r < 8; r++) {
    float p = pbc[r];
    p += __shfl_xor(p, 1); p += __shfl_xor(p, 2);
    p += __shfl_xor(p, 4); p += __shfl_xor(p, 8);
    pbc[r] = p;                          // bc for token m = r + 8*g
  }
  __syncthreads();

  // ---- dt GEMM: dt_raw[16x16 (K-pad 32)] @ dtproj[16x512] + softplus ------
  v16h adt;
#pragma unroll
  for (int i = 0; i < 16; i++) {
    int K = (i < 8 ? i : i + 8) + 8*g;
    adt[i] = (K < DT_RANK) ? s_dtr[m16 * 16 + K] : (_Float16)0.f;
  }
  const _Float16* pw_dt = pw + PW_DTPROJ_OFF;
  for (int nt = 0; nt < 32; nt++) {
    v8f acc; for (int r = 0; r < 8; r++) acc[r] = 0.f;
    v16h b = *(const v16h*)(pw_dt + (nt*32 + lane) * 16);
    acc = __builtin_amdgcn_wmma_f32_16x16x32_f16(false, adt, false, b,
                                                 (short)0, acc, false, false);
    int d = nt*16 + m16;
    float db = dt_b[d], dp = dpar[d];
    for (int r = 0; r < 8; r++) {
      float z  = acc[r] + db;
      float dt = (z > 20.f) ? z : log1pf(__expf(z));        // softplus
      int mrow = r + 8*g;
      float xc = (float)s_xc[mrow * D_INNER + d];
      float gv = (float)s_gy[mrow * D_INNER + d];
      float y  = xc * (dt * pbc[r] + dp);                   // + D-skip
      y *= silu_f(gv);                                      // * silu(gate)
      s_gy[mrow * D_INNER + d] = (_Float16)y;               // reuse gate slot
    }
  }
  __syncthreads();

  // ---- GEMM4: y[16x512] @ out_w[512x256] + out_b + residual ---------------
  const _Float16* pw_out = pw + PW_OUTW_OFF;
  for (int nt = 0; nt < 16; nt++) {
    v8f acc; for (int r = 0; r < 8; r++) acc[r] = 0.f;
    for (int kt = 0; kt < 16; kt++) {
      v16h a = loadA(s_gy, kt);
      v16h b = *(const v16h*)(pw_out + ((kt*16 + nt)*32 + lane) * 16);
      acc = __builtin_amdgcn_wmma_f32_16x16x32_f16(false, a, false, b,
                                                   (short)0, acc, false, false);
    }
    int col = nt*16 + m16;
    float ob = out_b[col];
    for (int r = 0; r < 8; r++) {
      int mrow = r + 8*g;
      float res = xin[(size_t)(tok0 + mrow) * D_MODEL + col];
      xout[(size_t)(tok0 + mrow) * D_MODEL + col] = acc[r] + ob + res;
    }
  }
}

// ---------------------------------------------------------------------------
extern "C" void kernel_launch(void* const* d_in, const int* in_sizes, int n_in,
                              void* d_out, int out_size, void* d_ws, size_t ws_size,
                              hipStream_t stream) {
  const float* x        = (const float*)d_in[0];
  const float* ln_w     = (const float*)d_in[1];
  const float* ln_b     = (const float*)d_in[2];
  const float* in_w     = (const float*)d_in[3];
  const float* in_b     = (const float*)d_in[4];
  const float* conv_w   = (const float*)d_in[5];
  const float* conv_b   = (const float*)d_in[6];
  const float* xproj_w  = (const float*)d_in[7];
  const float* dtproj_w = (const float*)d_in[8];
  const float* dtproj_b = (const float*)d_in[9];
  // d_in[10] = A_log: mathematically unused (h0 = 0 and sequence length 1)
  const float* D_param  = (const float*)d_in[11];
  const float* out_w    = (const float*)d_in[12];
  const float* out_b    = (const float*)d_in[13];

  float*    h_buf = (float*)d_ws;                                  // 3.28 MB
  _Float16* pw    = (_Float16*)((char*)d_ws +
                                (size_t)N_TOK * D_MODEL * sizeof(float));

  // 1) pack all weights -> f16 WMMA-B tile layout (both layers)
  int units = N_LAYERS * TILES_PER_LAYER * 32;
  pack_weights<<<(units + 255) / 256, 256, 0, stream>>>(
      in_w, xproj_w, dtproj_w, out_w, pw);

  // 2) layer 0: x -> h_buf
  ssm_layer<<<N_TOK / 16, 32, 0, stream>>>(
      x, h_buf, pw,
      ln_w, ln_b, in_b, conv_w, conv_b, dtproj_b, D_param, out_b);

  // 3) layer 1: h_buf -> d_out
  ssm_layer<<<N_TOK / 16, 32, 0, stream>>>(
      h_buf, (float*)d_out, pw + PW_LAYER_HALVES,
      ln_w + D_MODEL, ln_b + D_MODEL, in_b + 2*D_INNER,
      conv_w + 4*D_INNER, conv_b + D_INNER, dtproj_b + D_INNER,
      D_param + D_INNER, out_b + D_MODEL);
}
